// AttentiveFPPooling_72164040507935
// MI455X (gfx1250) — matrix operand order repaired and checked by reading, not compiled, not run on hardware
//
#include <hip/hip_runtime.h>
#include <hip/hip_bf16.h>

// ---------------------------------------------------------------------------
// AttentiveFP pooling, MI455X (gfx1250, wave32).
// Key algebraic rewrite: gat_out = segment_sum(scores*x) @ W_node
// -> removes the N x 512 x 512 GEMM and the 512MB h_neighbors array.
// HBM traffic ~1.5GB (3 passes over x) => ~66us roofline at 23.3 TB/s.
// Dense GEMMs: v_wmma_f32_16x16x32_bf16, 32x64 register tile per wave
// (8 WMMA / K-step, 4x A-fragment reuse, 2x B-fragment reuse).
// ---------------------------------------------------------------------------

#define N_NODES 262144
#define NGRAPH  2048
#define DIM     512
#define HID3    1536
#define OUTD    128

typedef __attribute__((ext_vector_type(16))) __bf16        v16bf;
typedef __attribute__((ext_vector_type(8)))  float         v8f;
typedef __attribute__((ext_vector_type(4)))  unsigned int  v4u;

// ---- helpers --------------------------------------------------------------

__device__ __forceinline__ unsigned short f2bf(float f) {
  unsigned u = __float_as_uint(f);
  unsigned r = u + 0x7FFFu + ((u >> 16) & 1u);   // round-to-nearest-even
  return (unsigned short)(r >> 16);
}

// order-preserving float<->uint encoding for atomicMax on floats
__device__ __forceinline__ unsigned encOrd(float f) {
  unsigned u = __float_as_uint(f);
  return (u & 0x80000000u) ? ~u : (u | 0x80000000u);
}
__device__ __forceinline__ float decOrd(unsigned e) {
  unsigned u = (e & 0x80000000u) ? (e & 0x7FFFFFFFu) : ~e;
  return __uint_as_float(u);
}

__device__ __forceinline__ float leakyf(float a)   { return a > 0.f ? a : 0.01f * a; }
__device__ __forceinline__ float sigmoidf(float x) { return 1.f / (1.f + __expf(-x)); }

__device__ __forceinline__ float waveSum(float p) {
  #pragma unroll
  for (int off = 16; off; off >>= 1) p += __shfl_xor(p, off, 32);
  return p;
}

__device__ __forceinline__ void atomAddF(float* p, float v) {
  // lowers to global_atomic_add_f32 (device scope, relaxed)
  __hip_atomic_fetch_add(p, v, __ATOMIC_RELAXED, __HIP_MEMORY_SCOPE_AGENT);
}

// ---- pass 1: hOut[b] = segment_sum(x), latt[i] = x_i . w_att_l ------------
// 128 threads x 4 dims = 512 dims; 64 sorted nodes per block; register
// accumulation with one global atomic flush per segment boundary.
__global__ void k_pass1(const float* __restrict__ x, const int* __restrict__ seg,
                        const float* __restrict__ wl, float* __restrict__ hOut,
                        float* __restrict__ latt) {
  __shared__ int   segS[64];
  __shared__ float lattS[64];
  const int tid = threadIdx.x;
  const int node0 = blockIdx.x * 64;
  if (tid < 64) { segS[tid] = seg[node0 + tid]; lattS[tid] = 0.f; }
  __syncthreads();
  const int d0 = tid * 4;
  const float4 w4 = *(const float4*)(wl + d0);
  float a0 = 0.f, a1 = 0.f, a2 = 0.f, a3 = 0.f;
  int cur = segS[0];
  const int lane = tid & 31;
  for (int ln = 0; ln < 64; ++ln) {
    __builtin_prefetch(x + (size_t)(node0 + ln + 6) * DIM + d0, 0, 0);
    const int sg = segS[ln];
    if (sg != cur) {
      float* dst = hOut + (size_t)cur * DIM + d0;
      atomAddF(dst + 0, a0); atomAddF(dst + 1, a1);
      atomAddF(dst + 2, a2); atomAddF(dst + 3, a3);
      a0 = a1 = a2 = a3 = 0.f; cur = sg;
    }
    const float4 v = *(const float4*)(x + (size_t)(node0 + ln) * DIM + d0);
    a0 += v.x; a1 += v.y; a2 += v.z; a3 += v.w;
    float p = v.x * w4.x + v.y * w4.y + v.z * w4.z + v.w * w4.w;
    p = waveSum(p);
    if (lane == 0) atomicAdd(&lattS[ln], p);   // ds_add_f32
  }
  float* dst = hOut + (size_t)cur * DIM + d0;
  atomAddF(dst + 0, a0); atomAddF(dst + 1, a1);
  atomAddF(dst + 2, a2); atomAddF(dst + 3, a3);
  __syncthreads();
  if (tid < 64) latt[node0 + tid] = lattS[tid];
}

// ---- right_att[b] = out[b] . w_att_r, one wave per graph ------------------
__global__ void k_right_att(const float* __restrict__ h, const float* __restrict__ wr,
                            float* __restrict__ ratt) {
  const int wave = blockIdx.x * (blockDim.x >> 5) + (threadIdx.x >> 5);
  const int lane = threadIdx.x & 31;
  if (wave >= NGRAPH) return;
  const float* row = h + (size_t)wave * DIM + lane * 16;
  const float* wv  = wr + lane * 16;
  float p = 0.f;
  #pragma unroll
  for (int i = 0; i < 16; i += 4) {
    float4 a = *(const float4*)(row + i);
    float4 b = *(const float4*)(wv + i);
    p += a.x * b.x + a.y * b.y + a.z * b.z + a.w * b.w;
  }
  p = waveSum(p);
  if (lane == 0) ratt[wave] = p;
}

// ---- per-graph softmax stats ---------------------------------------------
__global__ void k_attmax(const int* __restrict__ seg, const float* __restrict__ latt,
                         const float* __restrict__ ratt, unsigned* __restrict__ menc) {
  const int i = blockIdx.x * blockDim.x + threadIdx.x;
  const int b = seg[i];
  const float a = leakyf(latt[i] + ratt[b]);
  atomicMax(&menc[b], encOrd(a));
}

__global__ void k_attden(const int* __restrict__ seg, const float* __restrict__ latt,
                         const float* __restrict__ ratt, const unsigned* __restrict__ menc,
                         float* __restrict__ denom) {
  const int i = blockIdx.x * blockDim.x + threadIdx.x;
  const int b = seg[i];
  const float a = leakyf(latt[i] + ratt[b]);
  atomAddF(&denom[b], __expf(a - decOrd(menc[b])));
}

// ---- weighted segment sum: s[b] = sum_i softmax_i * x_i -------------------
__global__ void k_wsum(const float* __restrict__ x, const int* __restrict__ seg,
                       const float* __restrict__ latt, const float* __restrict__ ratt,
                       const unsigned* __restrict__ menc, const float* __restrict__ denom,
                       float* __restrict__ s) {
  __shared__ int   segS[64];
  __shared__ float wS[64];
  const int tid = threadIdx.x;
  const int node0 = blockIdx.x * 64;
  if (tid < 64) {
    const int n = node0 + tid;
    const int b = seg[n];
    segS[tid] = b;
    const float a = leakyf(latt[n] + ratt[b]);
    wS[tid] = __expf(a - decOrd(menc[b])) / denom[b];
  }
  __syncthreads();
  const int d0 = tid * 4;
  float a0 = 0.f, a1 = 0.f, a2 = 0.f, a3 = 0.f;
  int cur = segS[0];
  for (int ln = 0; ln < 64; ++ln) {
    __builtin_prefetch(x + (size_t)(node0 + ln + 6) * DIM + d0, 0, 0);
    const int sg = segS[ln];
    if (sg != cur) {
      float* dst = s + (size_t)cur * DIM + d0;
      atomAddF(dst + 0, a0); atomAddF(dst + 1, a1);
      atomAddF(dst + 2, a2); atomAddF(dst + 3, a3);
      a0 = a1 = a2 = a3 = 0.f; cur = sg;
    }
    const float wgt = wS[ln];
    const float4 v = *(const float4*)(x + (size_t)(node0 + ln) * DIM + d0);
    a0 += wgt * v.x; a1 += wgt * v.y; a2 += wgt * v.z; a3 += wgt * v.w;
  }
  float* dst = s + (size_t)cur * DIM + d0;
  atomAddF(dst + 0, a0); atomAddF(dst + 1, a1);
  atomAddF(dst + 2, a2); atomAddF(dst + 3, a3);
}

// ---- casts ---------------------------------------------------------------
__global__ void k_cast_bf(const float* __restrict__ src, unsigned short* __restrict__ dst,
                          int count) {
  const int i = blockIdx.x * blockDim.x + threadIdx.x;
  if (i < count) dst[i] = f2bf(src[i]);
}

// dst[n*rows + k] = bf16(src[k*cols + n]) : transpose to [Nout, K] row-major
__global__ void k_tcast(const float* __restrict__ src, unsigned short* __restrict__ dst,
                        int rows, int cols) {
  const int i = blockIdx.x * blockDim.x + threadIdx.x;
  if (i >= rows * cols) return;
  const int n = i / rows, k = i % rows;
  dst[i] = f2bf(src[(size_t)k * cols + n]);
}

// ---- bf16 WMMA GEMM: C[M,Nout] = A[M,K=512] * W[Nout,K]^T (+bias)(+act) ---
// One wave computes a 32(M) x 64(N) register tile: 2 A-frags x 4 B-frags
// -> 8 v_wmma_f32_16x16x32_bf16 per K-step, 4x A reuse / 2x B reuse.
union FragBF { v16bf v; v4u u[2]; };

__device__ __forceinline__ void loadFragA(FragBF& f, const unsigned short* row,
                                          int kk, int half) {
  // 16-bit A 16x32 layout: K runs [h*8, h*8+8) and [16+h*8, 16+h*8+8)
  f.u[0] = *(const v4u*)(row + kk + half * 8);
  f.u[1] = *(const v4u*)(row + kk + 16 + half * 8);
}
__device__ __forceinline__ void loadFragB(FragBF& f, const unsigned short* row,
                                          int kk, int half) {
  // B fragment: K runs [h*8, h*8+8) and [16+h*8, 16+h*8+8), n = lane&15
  f.u[0] = *(const v4u*)(row + kk + half * 8);
  f.u[1] = *(const v4u*)(row + kk + 16 + half * 8);
}

template <int ACT, int OUTBF>
__global__ void k_gemm(const unsigned short* __restrict__ A,
                       const unsigned short* __restrict__ W,
                       const float* __restrict__ bias,
                       float* __restrict__ Cf, unsigned short* __restrict__ Cb,
                       int M, int Nout) {
  const int K = DIM;
  const int wave = blockIdx.x * (blockDim.x >> 5) + (threadIdx.x >> 5);
  const int Nt = Nout >> 6;                   // 64-wide N blocks
  const int total = (M >> 5) * Nt;            // 32-tall M blocks
  if (wave >= total) return;                  // wave-uniform: EXEC stays all-ones
  const int mt = wave / Nt, nt = wave % Nt;
  const int lane = threadIdx.x & 31;
  const int r16 = lane & 15, half = lane >> 4;
  const unsigned short* aRow0 = A + (size_t)(mt * 32 + r16) * K;
  const unsigned short* aRow1 = aRow0 + (size_t)16 * K;
  const unsigned short* bRow0 = W + (size_t)(nt * 64 + r16) * K;
  const unsigned short* bRow1 = bRow0 + (size_t)16 * K;
  const unsigned short* bRow2 = bRow0 + (size_t)32 * K;
  const unsigned short* bRow3 = bRow0 + (size_t)48 * K;

  const v8f vz = {0.f, 0.f, 0.f, 0.f, 0.f, 0.f, 0.f, 0.f};
  v8f acc[8];                                 // acc[mi*4 + ni]
  #pragma unroll
  for (int i = 0; i < 8; ++i) acc[i] = vz;

  #pragma unroll 2
  for (int kk = 0; kk < K; kk += 32) {
    FragBF a0, a1, b0, b1, b2, b3;
    loadFragA(a0, aRow0, kk, half);
    loadFragA(a1, aRow1, kk, half);
    loadFragB(b0, bRow0, kk, half);
    loadFragB(b1, bRow1, kk, half);
    loadFragB(b2, bRow2, kk, half);
    loadFragB(b3, bRow3, kk, half);
    acc[0] = __builtin_amdgcn_wmma_f32_16x16x32_bf16(false, a0.v, false, b0.v, (short)0, acc[0], false, false);
    acc[1] = __builtin_amdgcn_wmma_f32_16x16x32_bf16(false, a0.v, false, b1.v, (short)0, acc[1], false, false);
    acc[2] = __builtin_amdgcn_wmma_f32_16x16x32_bf16(false, a0.v, false, b2.v, (short)0, acc[2], false, false);
    acc[3] = __builtin_amdgcn_wmma_f32_16x16x32_bf16(false, a0.v, false, b3.v, (short)0, acc[3], false, false);
    acc[4] = __builtin_amdgcn_wmma_f32_16x16x32_bf16(false, a1.v, false, b0.v, (short)0, acc[4], false, false);
    acc[5] = __builtin_amdgcn_wmma_f32_16x16x32_bf16(false, a1.v, false, b1.v, (short)0, acc[5], false, false);
    acc[6] = __builtin_amdgcn_wmma_f32_16x16x32_bf16(false, a1.v, false, b2.v, (short)0, acc[6], false, false);
    acc[7] = __builtin_amdgcn_wmma_f32_16x16x32_bf16(false, a1.v, false, b3.v, (short)0, acc[7], false, false);
  }

  #pragma unroll
  for (int mi = 0; mi < 2; ++mi) {
    #pragma unroll
    for (int ni = 0; ni < 4; ++ni) {
      const int n = nt * 64 + ni * 16 + r16;
      const float bs = bias ? bias[n] : 0.f;
      const v8f av = acc[mi * 4 + ni];
      #pragma unroll
      for (int r = 0; r < 8; ++r) {
        const int m = mt * 32 + mi * 16 + half * 8 + r;  // C/D layout: VGPR r, half sel
        float v = av[r] + bs;
        if (ACT == 1) v = v > 0.f ? v : (__expf(v) - 1.f);   // ELU
        if (OUTBF) Cb[(size_t)m * Nout + n] = f2bf(v);
        else       Cf[(size_t)m * Nout + n] = v;
      }
    }
  }
}

// ---- fused GRU gates + silu ----------------------------------------------
__global__ void k_gru(const float* __restrict__ gi, const float* __restrict__ gh,
                      float* __restrict__ h) {
  const int i = blockIdx.x * blockDim.x + threadIdx.x;  // over NGRAPH*DIM
  const int b = i >> 9, j = i & 511;
  const float* gib = gi + (size_t)b * HID3;
  const float* ghb = gh + (size_t)b * HID3;
  const float r  = sigmoidf(gib[j]        + ghb[j]);
  const float z  = sigmoidf(gib[512 + j]  + ghb[512 + j]);
  const float nn = tanhf(gib[1024 + j] + r * ghb[1024 + j]);
  const float hp = h[i];
  const float o  = (1.f - z) * nn + z * hp;
  h[i] = o * sigmoidf(o);                                 // silu
}

// ---------------------------------------------------------------------------

extern "C" void kernel_launch(void* const* d_in, const int* in_sizes, int n_in,
                              void* d_out, int out_size, void* d_ws, size_t ws_size,
                              hipStream_t stream) {
  (void)in_sizes; (void)n_in; (void)out_size; (void)ws_size;
  const float* x     = (const float*)d_in[0];
  const int*   seg   = (const int*)d_in[1];
  const float* wl    = (const float*)d_in[2];
  const float* wr    = (const float*)d_in[3];
  const float* Wnode = (const float*)d_in[4];
  const float* Wih   = (const float*)d_in[5];
  const float* Whh   = (const float*)d_in[6];
  const float* bih   = (const float*)d_in[7];
  const float* bhh   = (const float*)d_in[8];
  const float* Wlin  = (const float*)d_in[9];
  const float* blin  = (const float*)d_in[10];
  float* out = (float*)d_out;

  char* ws = (char*)d_ws;
  size_t off = 0;
  auto alloc = [&](size_t bytes) -> void* {
    void* p = ws + off;
    off = (off + bytes + 255) & ~(size_t)255;
    return p;
  };
  float*          hOut = (float*)alloc((size_t)NGRAPH * DIM * 4);
  float*          latt = (float*)alloc((size_t)N_NODES * 4);
  float*          ratt = (float*)alloc((size_t)NGRAPH * 4);
  unsigned*       menc = (unsigned*)alloc((size_t)NGRAPH * 4);
  float*          deno = (float*)alloc((size_t)NGRAPH * 4);
  float*          sseg = (float*)alloc((size_t)NGRAPH * DIM * 4);
  unsigned short* Abf  = (unsigned short*)alloc((size_t)NGRAPH * DIM * 2);
  unsigned short* Hbf  = (unsigned short*)alloc((size_t)NGRAPH * DIM * 2);
  unsigned short* Obf  = (unsigned short*)alloc((size_t)NGRAPH * DIM * 2);
  float*          gi   = (float*)alloc((size_t)NGRAPH * HID3 * 4);
  float*          gh   = (float*)alloc((size_t)NGRAPH * HID3 * 4);
  unsigned short* WtN  = (unsigned short*)alloc((size_t)DIM * DIM * 2);
  unsigned short* WtIH = (unsigned short*)alloc((size_t)HID3 * DIM * 2);
  unsigned short* WtHH = (unsigned short*)alloc((size_t)HID3 * DIM * 2);
  unsigned short* WtL  = (unsigned short*)alloc((size_t)OUTD * DIM * 2);

  auto gemmBlocks = [](int M, int Nout) { return ((M / 32) * (Nout / 64) + 7) / 8; };

  // weight prep (bf16, [Nout, K] row-major)
  k_cast_bf<<<(HID3 * DIM + 255) / 256, 256, 0, stream>>>(Wih, WtIH, HID3 * DIM);
  k_cast_bf<<<(HID3 * DIM + 255) / 256, 256, 0, stream>>>(Whh, WtHH, HID3 * DIM);
  k_tcast<<<(DIM * DIM + 255) / 256, 256, 0, stream>>>(Wnode, WtN, DIM, DIM);
  k_tcast<<<(DIM * OUTD + 255) / 256, 256, 0, stream>>>(Wlin, WtL, DIM, OUTD);

  // pass 1: initial hidden state + left attention
  hipMemsetAsync(hOut, 0, (size_t)NGRAPH * DIM * 4, stream);
  k_pass1<<<N_NODES / 64, 128, 0, stream>>>(x, seg, wl, hOut, latt);

  for (int t = 0; t < 2; ++t) {
    hipMemsetAsync(menc, 0, (size_t)NGRAPH * 4, stream);
    hipMemsetAsync(deno, 0, (size_t)NGRAPH * 4, stream);
    hipMemsetAsync(sseg, 0, (size_t)NGRAPH * DIM * 4, stream);
    k_right_att<<<NGRAPH / 8, 256, 0, stream>>>(hOut, wr, ratt);
    k_attmax<<<N_NODES / 256, 256, 0, stream>>>(seg, latt, ratt, menc);
    k_attden<<<N_NODES / 256, 256, 0, stream>>>(seg, latt, ratt, menc, deno);
    k_wsum<<<N_NODES / 64, 128, 0, stream>>>(x, seg, latt, ratt, menc, deno, sseg);

    // h = elu(s @ W_node)  (bf16 out, feeds next GEMM's A)
    k_cast_bf<<<(NGRAPH * DIM + 255) / 256, 256, 0, stream>>>(sseg, Abf, NGRAPH * DIM);
    k_gemm<1, 1><<<gemmBlocks(NGRAPH, DIM), 256, 0, stream>>>(
        Abf, WtN, nullptr, nullptr, Hbf, NGRAPH, DIM);

    // gi = h @ W_ih^T + b_ih ; gh = h_prev @ W_hh^T + b_hh
    k_cast_bf<<<(NGRAPH * DIM + 255) / 256, 256, 0, stream>>>(hOut, Obf, NGRAPH * DIM);
    k_gemm<0, 0><<<gemmBlocks(NGRAPH, HID3), 256, 0, stream>>>(
        Hbf, WtIH, bih, gi, nullptr, NGRAPH, HID3);
    k_gemm<0, 0><<<gemmBlocks(NGRAPH, HID3), 256, 0, stream>>>(
        Obf, WtHH, bhh, gh, nullptr, NGRAPH, HID3);

    // out = silu(gru(h, h_prev))
    k_gru<<<NGRAPH * DIM / 256, 256, 0, stream>>>(gi, gh, hOut);
  }

  // final projection: out @ W_lin + b_lin -> d_out [2048,128]
  k_cast_bf<<<(NGRAPH * DIM + 255) / 256, 256, 0, stream>>>(hOut, Obf, NGRAPH * DIM);
  k_gemm<0, 0><<<gemmBlocks(NGRAPH, OUTD), 256, 0, stream>>>(
      Obf, WtL, blin, out, nullptr, NGRAPH, OUTD);
}
